// Model_24318104829969
// MI455X (gfx1250) — matrix operand order repaired and checked by reading, not compiled, not run on hardware
//
#include <hip/hip_runtime.h>
#include <hip/hip_bf16.h>

// ---------------- problem constants (match reference) ----------------
#define NNODES  10000
#define KSLOT   32
#define DIM     128
#define NHEAD   4
#define DH      32
#define NLAYER  2
#define NSTRUCT 200

typedef __attribute__((ext_vector_type(16))) _Float16 v16h;
typedef __attribute__((ext_vector_type(8)))  float    v8f;

__device__ __forceinline__ float gelu_exact(float x) {
  return 0.5f * x * (1.0f + erff(x * 0.70710678118f));
}

// ----------------------------------------------------------------------------
// wave-level GEMM: C(32xN) = A(32xK,LDS f16 row-major) @ B(KxN)
//   BMODE 0: B pre-packed in WMMA tile-major form (global): per (kt,nt) tile,
//            512 halves laid out so lane L reads 16 contiguous halves at L*16.
//   BMODE 1: B transposed in LDS: logical B[k][n] = Bt[n*ldbt + k] (contiguous).
// Output: f32 LDS (= or +=) and/or f16 LDS, optional bias / scale / exact GELU.
// m-tiles fused: one B load feeds two WMMAs.
// ----------------------------------------------------------------------------
template <int BMODE>
__device__ __forceinline__ void wave_gemm(
    const _Float16* __restrict__ A, int lda,
    const _Float16* __restrict__ B, int bstride,  // BMODE0: NT=N/16; BMODE1: ldbt
    int N, int K, const float* __restrict__ bias,
    float* __restrict__ Cf, int ldcf, bool acc,
    _Float16* __restrict__ Ch, int ldch, bool dogelu,
    float scale, int lane) {
  const int nn   = lane & 15;
  const int khalf = (lane >> 4);            // 0/1: which K half-group this lane holds
  const int mhi  = khalf << 3;
  const _Float16* Arow0 = A + (size_t)nn * lda + (khalf << 4);
  const _Float16* Arow1 = Arow0 + (size_t)16 * lda;

#pragma clang loop unroll(disable)
  for (int n0 = 0; n0 < N; n0 += 16) {
    float bv = bias ? bias[n0 + nn] : 0.0f;
    v8f c0, c1;
#pragma unroll
    for (int r = 0; r < 8; ++r) { c0[r] = bv; c1[r] = bv; }

#pragma clang loop unroll(disable)
    for (int k0 = 0; k0 < K; k0 += 32) {
      v16h b;
      if (BMODE == 0) {
        const _Float16* p =
            B + (((size_t)(k0 >> 5) * bstride + (n0 >> 4)) << 9) + (lane << 4);
        b = *(const v16h*)p;                       // 32B contiguous per lane
      } else {
        const _Float16* p = B + (size_t)(n0 + nn) * bstride + k0 + (khalf << 4);
        b = *(const v16h*)p;                       // 32B contiguous per lane (LDS)
      }
      v16h a0, a1;
      const _Float16* pa0 = Arow0 + k0;
      const _Float16* pa1 = Arow1 + k0;
#pragma unroll
      for (int i = 0; i < 8; ++i) {
        a0[i] = pa0[i]; a0[8 + i] = pa0[8 + i];
        a1[i] = pa1[i]; a1[8 + i] = pa1[8 + i];
      }
      c0 = __builtin_amdgcn_wmma_f32_16x16x32_f16(false, a0, false, b,
                                                  (short)0, c0, false, false);
      c1 = __builtin_amdgcn_wmma_f32_16x16x32_f16(false, a1, false, b,
                                                  (short)0, c1, false, false);
    }

#pragma unroll
    for (int r = 0; r < 8; ++r) {
      int m0r = mhi + r, m1r = 16 + mhi + r;
      float v0 = c0[r] * scale, v1 = c1[r] * scale;
      if (dogelu) { v0 = gelu_exact(v0); v1 = gelu_exact(v1); }
      if (Cf) {
        float* q0 = Cf + (size_t)m0r * ldcf + n0 + nn;
        float* q1 = Cf + (size_t)m1r * ldcf + n0 + nn;
        if (acc) { *q0 += v0; *q1 += v1; } else { *q0 = v0; *q1 = v1; }
      }
      if (Ch) {
        Ch[(size_t)m0r * ldch + n0 + nn] = (_Float16)v0;
        Ch[(size_t)m1r * ldch + n0 + nn] = (_Float16)v1;
      }
    }
  }
}

__device__ __forceinline__ int sp2idx(int s) {
  return s == 1 ? 0 : (s == 6 ? 1 : (s == 7 ? 2 : 3));
}

__device__ __forceinline__ void layernorm_row(const float* x, _Float16* h,
                                              const float* g, const float* b,
                                              int lane) {
  const float* row = x + (size_t)lane * DIM;
  float mu = 0.f;
  for (int d = 0; d < DIM; ++d) mu += row[d];
  mu *= (1.0f / DIM);
  float var = 0.f;
  for (int d = 0; d < DIM; ++d) { float t = row[d] - mu; var += t * t; }
  var *= (1.0f / DIM);
  float inv = rsqrtf(var + 1e-5f);
  _Float16* hr = h + (size_t)lane * DIM;
  for (int d = 0; d < DIM; ++d)
    hr[d] = (_Float16)((row[d] - mu) * inv * g[d] + b[d]);
}

// ---------------- helper kernels ----------------
__global__ void zero_f32(float* p, int n) {
  int i = blockIdx.x * blockDim.x + threadIdx.x;
  if (i < n) p[i] = 0.0f;
}
__global__ void fill_i32(int* p, int n, int v) {
  int i = blockIdx.x * blockDim.x + threadIdx.x;
  if (i < n) p[i] = v;
}
__global__ void seg_first(const int* __restrict__ centers, int* __restrict__ first, int E) {
  int e = blockIdx.x * blockDim.x + threadIdx.x;
  if (e < E) atomicMin(&first[centers[e]], e);
}
__global__ void build_nef(const int* __restrict__ centers, const int* __restrict__ first,
                          int* __restrict__ nef, int E) {
  int e = blockIdx.x * blockDim.x + threadIdx.x;
  if (e < E) {
    int c = centers[e];
    int pos = e - first[c];
    if (pos >= 0 && pos < KSLOT) nef[c * KSLOT + pos] = e;
  }
}
// Pack f32 W(KxN, row-major) into f16 WMMA-B tile-major layout:
// out[((kt*NT+nt)*32 + lane)*16 + i] = W[(kt*32 + (lane>>4)*16 + i)*N + nt*16 + (lane&15)]
__global__ void pack_b16(const float* __restrict__ W, _Float16* __restrict__ P,
                         int K, int N) {
  int tid = blockIdx.x * blockDim.x + threadIdx.x;
  if (tid >= K * N) return;
  int i    = tid & 15;
  int lane = (tid >> 4) & 31;
  int tile = tid >> 9;
  int NT   = N >> 4;
  int nt   = tile % NT;
  int kt   = tile / NT;
  int k    = (kt << 5) + ((lane >> 4) << 4) + i;
  int n    = (nt << 4) + (lane & 15);
  P[tid]   = (_Float16)W[(size_t)k * N + n];
}

// ---------------- main per-node transformer: 1 wave per node ----------------
__launch_bounds__(32)
__global__ void node_former(
    const int* __restrict__ nef, const int* __restrict__ neighbors,
    const int* __restrict__ species, const int* __restrict__ segidx,
    const float* __restrict__ edge_vectors,
    const float* __restrict__ W_cart, const float* __restrict__ b_cart,
    const float* __restrict__ emb_c, const float* __restrict__ emb_n,
    const _Float16* __restrict__ Wcomp16, const float* __restrict__ b_comp,
    const float* __restrict__ ln1_g, const float* __restrict__ ln1_b,
    const _Float16* __restrict__ Wqkv16, const float* __restrict__ bqkv,
    const _Float16* __restrict__ Wo16, const float* __restrict__ bo,
    const float* __restrict__ ln2_g, const float* __restrict__ ln2_b,
    const _Float16* __restrict__ W116, const float* __restrict__ b1,
    const _Float16* __restrict__ W216, const float* __restrict__ b2,
    const float* __restrict__ W_read, const float* __restrict__ comp_w,
    float* __restrict__ out) {
  const int n    = blockIdx.x;
  const int lane = threadIdx.x;

  __shared__ alignas(32) float    xf[32 * 128];   // residual stream (f32)
  __shared__ alignas(32) _Float16 ah[32 * 128];   // LN out / concat head outputs
  __shared__ alignas(32) _Float16 big[32 * 512];  // feat384 / qkv384 / mlp hidden
  __shared__ alignas(32) _Float16 vT[128 * 32];   // V transposed: vT[d][key]
  __shared__ alignas(32) float    sc[32 * 32];    // attention scores
  __shared__ alignas(32) _Float16 at16[32 * 32];  // softmaxed+masked attention
  __shared__ float rm[32];
  __shared__ float red[32];

  // ---- per-slot edge gather + encoder features ----
  int e = nef[n * KSLOT + lane];
  float ev0 = 0.f, ev1 = 0.f, ev2 = 0.f, rmask = 0.f;
  int elc = 1, eln = 1;                 // reference NEF fill quirk (fill = species 1)
  if (e >= 0) {
    ev0 = edge_vectors[3 * e + 0];
    ev1 = edge_vectors[3 * e + 1];
    ev2 = edge_vectors[3 * e + 2];
    float r = sqrtf(ev0 * ev0 + ev1 * ev1 + ev2 * ev2);
    rmask = (r < 3.0f) ? 1.0f
                       : 0.5f * (__cosf(3.14159265358979f * (r - 3.0f) * 0.5f) + 1.0f);
    elc = sp2idx(species[n]);
    eln = sp2idx(species[neighbors[e]]);
  }
  rm[lane] = rmask;
  for (int d = 0; d < DIM; ++d) {
    float cart = ev0 * W_cart[d] + ev1 * W_cart[DIM + d] + ev2 * W_cart[2 * DIM + d] + b_cart[d];
    big[(size_t)lane * 384 + d]           = (_Float16)cart;
    big[(size_t)lane * 384 + DIM + d]     = (_Float16)emb_c[elc * DIM + d];
    big[(size_t)lane * 384 + 2 * DIM + d] = (_Float16)emb_n[eln * DIM + d];
  }
  __syncthreads();

  // x = feat(32x384) @ W_comp + b_comp
  wave_gemm<0>(big, 384, Wcomp16, 8, 128, 384, b_comp,
               xf, 128, false, nullptr, 0, false, 1.0f, lane);
  __syncthreads();

  for (int l = 0; l < NLAYER; ++l) {
    // LN1 -> ah (f16)
    layernorm_row(xf, ah, ln1_g + l * DIM, ln1_b + l * DIM, lane);
    __syncthreads();
    // qkv = h @ Wqkv + b  -> big (32x384 f16)
    wave_gemm<0>(ah, 128, Wqkv16 + (size_t)l * 128 * 384, 24, 384, 128,
                 bqkv + l * 384, nullptr, 0, false, big, 384, false, 1.0f, lane);
    __syncthreads();
    // V^T into LDS so att@V gets contiguous Bt loads (lane = key index)
    for (int d = 0; d < DIM; ++d)
      vT[(size_t)d * 32 + lane] = big[(size_t)lane * 384 + 2 * DIM + d];
    __syncthreads();

    for (int h = 0; h < NHEAD; ++h) {
      // S = (Q_h @ K_h^T) / sqrt(dh); K rows are contiguous -> Bt loader
      wave_gemm<1>(big + h * DH, 384, big + DIM + h * DH, 384, 32, 32, nullptr,
                   sc, 32, false, nullptr, 0, false, 0.17677669529f, lane);
      __syncthreads();
      // softmax over row `lane`, then post-softmax radial mask
      {
        float mx = -1e30f;
        for (int j = 0; j < 32; ++j) mx = fmaxf(mx, sc[lane * 32 + j]);
        float s = 0.f;
        for (int j = 0; j < 32; ++j) s += __expf(sc[lane * 32 + j] - mx);
        float inv = 1.0f / s;
        for (int j = 0; j < 32; ++j)
          at16[lane * 32 + j] = (_Float16)(__expf(sc[lane * 32 + j] - mx) * inv * rm[j]);
      }
      __syncthreads();
      // O_h = att @ V_h -> ah[:, h*32 .. h*32+31]; Bt = vT (contiguous)
      wave_gemm<1>(at16, 32, vT + (size_t)h * DH * 32, 32, 32, 32, nullptr,
                   nullptr, 0, false, ah + h * DH, 128, false, 1.0f, lane);
      __syncthreads();
    }

    // x += O @ Wo + bo
    wave_gemm<0>(ah, 128, Wo16 + (size_t)l * 128 * 128, 8, 128, 128,
                 bo + l * DIM, xf, 128, true, nullptr, 0, false, 1.0f, lane);
    __syncthreads();
    // LN2 -> ah
    layernorm_row(xf, ah, ln2_g + l * DIM, ln2_b + l * DIM, lane);
    __syncthreads();
    // hidden = gelu(h2 @ W1 + b1) -> big (32x512 f16)
    wave_gemm<0>(ah, 128, W116 + (size_t)l * 128 * 512, 32, 512, 128,
                 b1 + l * 512, nullptr, 0, false, big, 512, true, 1.0f, lane);
    __syncthreads();
    // x += hidden @ W2 + b2
    wave_gemm<0>(big, 512, W216 + (size_t)l * 512 * 128, 8, 128, 512,
                 b2 + l * DIM, xf, 128, true, nullptr, 0, false, 1.0f, lane);
    __syncthreads();
  }

  // readout: edge energies, node sum, segment scatter (+ composition term)
  float s = 0.f;
  for (int d = 0; d < DIM; ++d) s += xf[(size_t)lane * DIM + d] * W_read[d];
  red[lane] = s * rm[lane];
  __syncthreads();
  if (lane == 0) {
    float a = 0.f;
    for (int j = 0; j < 32; ++j) a += red[j];
    a += comp_w[sp2idx(species[n])];
    atomicAdd(&out[segidx[n]], a);
  }
}

// ---------------- launch ----------------
extern "C" void kernel_launch(void* const* d_in, const int* in_sizes, int n_in,
                              void* d_out, int out_size, void* d_ws, size_t ws_size,
                              hipStream_t stream) {
  const float* edge_vectors = (const float*)d_in[0];
  const int*   centers      = (const int*)d_in[1];
  const int*   neighbors    = (const int*)d_in[2];
  const int*   species      = (const int*)d_in[3];
  const int*   segidx       = (const int*)d_in[4];
  const float* W_cart       = (const float*)d_in[5];
  const float* b_cart       = (const float*)d_in[6];
  const float* emb_c        = (const float*)d_in[7];
  const float* emb_n        = (const float*)d_in[8];
  const float* W_comp       = (const float*)d_in[9];
  const float* b_comp       = (const float*)d_in[10];
  const float* ln1_g        = (const float*)d_in[11];
  const float* ln1_b        = (const float*)d_in[12];
  const float* Wqkv         = (const float*)d_in[13];
  const float* bqkv         = (const float*)d_in[14];
  const float* Wo           = (const float*)d_in[15];
  const float* bo           = (const float*)d_in[16];
  const float* ln2_g        = (const float*)d_in[17];
  const float* ln2_b        = (const float*)d_in[18];
  const float* W1           = (const float*)d_in[19];
  const float* b1           = (const float*)d_in[20];
  const float* W2           = (const float*)d_in[21];
  const float* b2           = (const float*)d_in[22];
  const float* W_read       = (const float*)d_in[23];
  const float* comp_w       = (const float*)d_in[24];
  float* out = (float*)d_out;

  const int E = in_sizes[1];

  // workspace carving (256B aligned)
  char*  ws  = (char*)d_ws;
  size_t off = 0;
  auto carve = [&](size_t bytes) -> void* {
    void* p = ws + off;
    off = (off + bytes + 255) & ~(size_t)255;
    return p;
  };
  int*      first   = (int*)carve((size_t)NNODES * 4);
  int*      nef     = (int*)carve((size_t)NNODES * KSLOT * 4);
  _Float16* Wcomp16 = (_Float16*)carve((size_t)384 * 128 * 2);
  _Float16* Wqkv16  = (_Float16*)carve((size_t)2 * 128 * 384 * 2);
  _Float16* Wo16    = (_Float16*)carve((size_t)2 * 128 * 128 * 2);
  _Float16* W116    = (_Float16*)carve((size_t)2 * 128 * 512 * 2);
  _Float16* W216    = (_Float16*)carve((size_t)2 * 512 * 128 * 2);
  (void)n_in; (void)ws_size; (void)out_size;

  const int B = 256;
  zero_f32<<<(NSTRUCT + B - 1) / B, B, 0, stream>>>(out, NSTRUCT);
  fill_i32<<<(NNODES + B - 1) / B, B, 0, stream>>>(first, NNODES, 0x7fffffff);
  fill_i32<<<(NNODES * KSLOT + B - 1) / B, B, 0, stream>>>(nef, NNODES * KSLOT, -1);
  seg_first<<<(E + B - 1) / B, B, 0, stream>>>(centers, first, E);
  build_nef<<<(E + B - 1) / B, B, 0, stream>>>(centers, first, nef, E);

  // pack weights to f16 WMMA-B tile-major layout (per layer for stacked weights)
  pack_b16<<<(384 * 128 + B - 1) / B, B, 0, stream>>>(W_comp, Wcomp16, 384, 128);
  for (int l = 0; l < NLAYER; ++l) {
    pack_b16<<<(128 * 384 + B - 1) / B, B, 0, stream>>>(
        Wqkv + (size_t)l * 128 * 384, Wqkv16 + (size_t)l * 128 * 384, 128, 384);
    pack_b16<<<(128 * 128 + B - 1) / B, B, 0, stream>>>(
        Wo + (size_t)l * 128 * 128, Wo16 + (size_t)l * 128 * 128, 128, 128);
    pack_b16<<<(128 * 512 + B - 1) / B, B, 0, stream>>>(
        W1 + (size_t)l * 128 * 512, W116 + (size_t)l * 128 * 512, 128, 512);
    pack_b16<<<(512 * 128 + B - 1) / B, B, 0, stream>>>(
        W2 + (size_t)l * 512 * 128, W216 + (size_t)l * 512 * 128, 512, 128);
  }

  node_former<<<NNODES, 32, 0, stream>>>(
      nef, neighbors, species, segidx, edge_vectors,
      W_cart, b_cart, emb_c, emb_n,
      Wcomp16, b_comp, ln1_g, ln1_b,
      Wqkv16, bqkv, Wo16, bo, ln2_g, ln2_b,
      W116, b1, W216, b2, W_read, comp_w, out);
}
